// StandardizedMamba_81329500717467
// MI455X (gfx1250) — matrix operand order repaired and checked by reading, not compiled, not run on hardware
//
#include <hip/hip_runtime.h>
#include <hip/hip_bf16.h>
#include <math.h>

typedef __bf16 bf16;
typedef __attribute__((ext_vector_type(8)))  bf16  v8bf;
typedef __attribute__((ext_vector_type(16))) bf16  v16bf;
typedef __attribute__((ext_vector_type(8)))  float v8f;

constexpr int VOCAB = 32000;
constexpr int DMODEL = 512;
constexpr int NLAYER = 4;
constexpr int DSTATE = 16;
constexpr int DINNER = 1024;
constexpr int BATCH  = 2;
constexpr int SEQ    = 2048;
constexpr int NT     = BATCH * SEQ;   // 4096 token rows

__device__ __forceinline__ float sigmoid_f(float x) { return 1.0f / (1.0f + expf(-x)); }

// ---------------------------------------------------------------------------
// WMMA GEMM: C[M,N] = A[M,K] * Bt[N,K]^T   (A,Bt bf16 k-contiguous, C fp32)
// Block = 128 threads = 4 waves. Each wave computes a 32(M) x 64(N) tile:
// per K-step, 2 A frags + 4 B frags feed 8 v_wmma_f32_16x16x32_bf16, giving
// 2.7x better register-level reuse than one-tile-per-wave (24B/lane per WMMA).
// Fragment layouts per CDNA5 ISA 7.12.2 (16-bit A 16x32, B 32x16, f32 C/D).
// ---------------------------------------------------------------------------
__global__ void wmma_gemm_bf16(const bf16* __restrict__ A, const bf16* __restrict__ Bt,
                               float* __restrict__ C, int M, int N, int K)
{
    const int wave   = threadIdx.x >> 5;
    const int lane   = threadIdx.x & 31;
    const int half   = lane >> 4;           // 0: lanes 0-15, 1: lanes 16-31
    const int lm     = lane & 15;
    const int tileM  = blockIdx.y * 32;
    const int tileN0 = blockIdx.x * 256 + wave * 64;
    if (tileN0 >= N) return;                // wave-uniform: EXEC stays all-1s

    // A rows for the two M subtiles:
    // lanes0-15 carry K {0..7,16..23}, lanes16-31 carry K {8..15,24..31}
    const bf16* arow0 = A + (size_t)(tileM + lm) * K + half * 8;
    const bf16* arow1 = arow0 + (size_t)16 * K;

    // B columns (rows of Bt) for the four N subtiles; clamp index for partial
    // tiles so loads stay in-bounds (garbage subtiles are dropped at store).
    const bf16* brow[4];
#pragma unroll
    for (int j = 0; j < 4; ++j) {
        int n = tileN0 + j * 16 + lm;
        if (n > N - 1) n = N - 1;
        brow[j] = Bt + (size_t)n * K + half * 8;
    }

    v8f acc[2][4] = {};
    for (int kb = 0; kb < K; kb += 32) {
        union { v16bf v; v8bf h[2]; } a0, a1, b[4];
        a0.h[0] = *(const v8bf*)(arow0 + kb);        // K kb+{0..7} / {8..15}
        a0.h[1] = *(const v8bf*)(arow0 + kb + 16);   // K kb+{16..23} / {24..31}
        a1.h[0] = *(const v8bf*)(arow1 + kb);
        a1.h[1] = *(const v8bf*)(arow1 + kb + 16);
#pragma unroll
        for (int j = 0; j < 4; ++j) {
            b[j].h[0] = *(const v8bf*)(brow[j] + kb);      // K kb+{0..7} / {8..15}
            b[j].h[1] = *(const v8bf*)(brow[j] + kb + 16); // K kb+{16..23} / {24..31}
        }
#pragma unroll
        for (int j = 0; j < 4; ++j) {
            acc[0][j] = __builtin_amdgcn_wmma_f32_16x16x32_bf16(
                false, a0.v, false, b[j].v, (short)0, acc[0][j], false, false);
            acc[1][j] = __builtin_amdgcn_wmma_f32_16x16x32_bf16(
                false, a1.v, false, b[j].v, (short)0, acc[1][j], false, false);
        }
    }

    // D layout: lane<16 -> rows +0..7, lane>=16 -> rows +8..15, col = subtileN+lm
#pragma unroll
    for (int i = 0; i < 2; ++i) {
        const int mbase = tileM + i * 16 + half * 8;
#pragma unroll
        for (int j = 0; j < 4; ++j) {
            const int n = tileN0 + j * 16;
            if (n < N) {
                float* crow = C + (size_t)mbase * N + n + lm;
#pragma unroll
                for (int v = 0; v < 8; ++v) crow[(size_t)v * N] = acc[i][j][v];
            }
        }
    }
}

// ---------------------------------------------------------------------------
// Weight transpose + convert: Wt[n*K+k] = (bf16) W[k*N+n]
// ---------------------------------------------------------------------------
__global__ void transpose_conv_kernel(const float* __restrict__ W, bf16* __restrict__ Wt,
                                      int K, int N)
{
    size_t total = (size_t)K * N;
    for (size_t i = (size_t)blockIdx.x * blockDim.x + threadIdx.x; i < total;
         i += (size_t)gridDim.x * blockDim.x) {
        size_t n = i / (size_t)K, k = i - n * (size_t)K;
        Wt[i] = (bf16)W[k * (size_t)N + n];
    }
}

// Strided f32 -> bf16 convert: dst[r*cols+c] = src[r*srcStride+c]
__global__ void conv_strided_kernel(const float* __restrict__ src, bf16* __restrict__ dst,
                                    int srcStride, int cols, size_t total)
{
    for (size_t i = (size_t)blockIdx.x * blockDim.x + threadIdx.x; i < total;
         i += (size_t)gridDim.x * blockDim.x) {
        size_t r = i / (size_t)cols, c = i - r * (size_t)cols;
        dst[i] = (bf16)src[r * (size_t)srcStride + c];
    }
}

// Embedding gather: x[r,:] = emb[ids[r],:]  (fp32 + bf16 copies)
__global__ void embed_kernel(const int* __restrict__ ids, const float* __restrict__ emb,
                             float* __restrict__ x, bf16* __restrict__ xbf)
{
    size_t i = (size_t)blockIdx.x * blockDim.x + threadIdx.x;
    if (i >= (size_t)NT * DMODEL) return;
    size_t r = i >> 9, c = i & 511;
    float v = emb[(size_t)ids[r] * DMODEL + c];
    x[i] = v;  xbf[i] = (bf16)v;
}

// Split xp -> silu(x_ssm) bf16 (for W_x GEMM) and silu(z) fp32 (for gating)
__global__ void split_silu_kernel(const float* __restrict__ xp,
                                  bf16* __restrict__ sxbf, float* __restrict__ z)
{
    size_t i = (size_t)blockIdx.x * blockDim.x + threadIdx.x;
    if (i >= (size_t)NT * DINNER) return;
    size_t r = i >> 10, j = i & 1023;
    float xs = xp[r * (size_t)(2 * DINNER) + j];
    float zr = xp[r * (size_t)(2 * DINNER) + DINNER + j];
    sxbf[i] = (bf16)(xs * sigmoid_f(xs));
    z[i]    = zr * sigmoid_f(zr);
}

// dt = softplus(dtraw + b_dt), in place
__global__ void softplus_bias_kernel(float* __restrict__ dtraw, const float* __restrict__ b_dt)
{
    size_t i = (size_t)blockIdx.x * blockDim.x + threadIdx.x;
    if (i >= (size_t)NT * DINNER) return;
    float v = dtraw[i] + b_dt[i & (DINNER - 1)];
    dtraw[i] = (v > 20.0f) ? v : log1pf(expf(v));
}

// ---------------------------------------------------------------------------
// Selective scan: one thread per (batch, channel); h[16] in registers;
// B_t / C_t staged in LDS once per timestep. Grid = (DINNER/256, BATCH).
// y = (sum_s C*h + D*u) * silu(z), written as bf16 for the W_out GEMM.
// ---------------------------------------------------------------------------
__global__ void scan_kernel(const float* __restrict__ xp,   // (B,S,2048): u = xp[...,:1024]
                            const float* __restrict__ sp,   // (B,S,1056): B@1024, C@1040
                            const float* __restrict__ dt,   // (B,S,1024)
                            const float* __restrict__ z,    // (B,S,1024) silu'd
                            const float* __restrict__ A_log,// (1024,16) this layer
                            const float* __restrict__ Dp_,  // (1024)
                            bf16* __restrict__ ybf)         // (B,S,1024)
{
    const int b  = blockIdx.y;
    const int di = blockIdx.x * blockDim.x + threadIdx.x;

    float a[DSTATE], h[DSTATE];
#pragma unroll
    for (int s = 0; s < DSTATE; ++s) { a[s] = -expf(A_log[di * DSTATE + s]); h[s] = 0.0f; }
    const float Dp = Dp_[di];

    __shared__ float sB[DSTATE], sC[DSTATE];

    for (int t = 0; t < SEQ; ++t) {
        size_t row = (size_t)b * SEQ + t;
        if (threadIdx.x < 2 * DSTATE) {
            float v = sp[row * 1056 + DINNER + threadIdx.x];
            if (threadIdx.x < DSTATE) sB[threadIdx.x] = v;
            else                      sC[threadIdx.x - DSTATE] = v;
        }
        __syncthreads();

        float u = xp[row * (size_t)(2 * DINNER) + di];
        float d = dt[row * (size_t)DINNER + di];
        float y = 0.0f;
#pragma unroll
        for (int s = 0; s < DSTATE; ++s) {
            float Ad = 1.0f + a[s] * d;
            Ad = fminf(fmaxf(Ad, 0.0f), 1.0f);
            h[s] = Ad * h[s] + d * sB[s] * u;
            y += sC[s] * h[s];
        }
        y = (y + Dp * u) * z[row * (size_t)DINNER + di];
        ybf[row * (size_t)DINNER + di] = (bf16)y;
        __syncthreads();
    }
}

// ---------------------------------------------------------------------------
// LayerNorm over D=512. mode==1: x_new = x + LN(out + x)  (layer epilogue)
//                       mode==0: x_new = LN(x)            (final norm)
// Writes fp32 (residual stream) and bf16 (next GEMM operand).
// ---------------------------------------------------------------------------
__global__ void layernorm_kernel(const float* __restrict__ inp, float* __restrict__ x,
                                 bf16* __restrict__ xbf,
                                 const float* __restrict__ w, const float* __restrict__ bias,
                                 int mode)
{
    const int r = blockIdx.x, tid = threadIdx.x;
    size_t base = (size_t)r * DMODEL;
    float x0 = x[base + tid], x1 = x[base + tid + 256];
    float t0, t1;
    if (mode) { t0 = inp[base + tid] + x0; t1 = inp[base + tid + 256] + x1; }
    else      { t0 = x0;                   t1 = x1; }

    __shared__ float ssum[256], ssq[256];
    ssum[tid] = t0 + t1;
    ssq[tid]  = t0 * t0 + t1 * t1;
    __syncthreads();
    for (int o = 128; o > 0; o >>= 1) {
        if (tid < o) { ssum[tid] += ssum[tid + o]; ssq[tid] += ssq[tid + o]; }
        __syncthreads();
    }
    float mean = ssum[0] * (1.0f / DMODEL);
    float var  = ssq[0] * (1.0f / DMODEL) - mean * mean;
    float rstd = rsqrtf(var + 1e-5f);

    float l0 = (t0 - mean) * rstd * w[tid]       + bias[tid];
    float l1 = (t1 - mean) * rstd * w[tid + 256] + bias[tid + 256];
    float n0 = mode ? (x0 + l0) : l0;
    float n1 = mode ? (x1 + l1) : l1;
    x[base + tid] = n0;          x[base + tid + 256] = n1;
    xbf[base + tid] = (bf16)n0;  xbf[base + tid + 256] = (bf16)n1;
}

// ---------------------------------------------------------------------------
extern "C" void kernel_launch(void* const* d_in, const int* in_sizes, int n_in,
                              void* d_out, int out_size, void* d_ws, size_t ws_size,
                              hipStream_t stream)
{
    const int*   ids   = (const int*)  d_in[0];
    const float* emb   = (const float*)d_in[1];
    const float* W_in  = (const float*)d_in[2];
    const float* W_x   = (const float*)d_in[3];
    const float* W_dt  = (const float*)d_in[4];
    const float* b_dt  = (const float*)d_in[5];
    const float* A_log = (const float*)d_in[6];
    const float* D_par = (const float*)d_in[7];
    const float* W_out = (const float*)d_in[8];
    const float* ln_w  = (const float*)d_in[9];
    const float* ln_b  = (const float*)d_in[10];
    const float* fln_w = (const float*)d_in[11];
    const float* fln_b = (const float*)d_in[12];
    (void)in_sizes; (void)n_in; (void)out_size; (void)ws_size;

    char* ws = (char*)d_ws;
    size_t off = 0;
    auto alloc = [&](size_t bytes) -> void* {
        void* p = ws + off;
        off += (bytes + 255) & ~(size_t)255;
        return p;
    };

    float* x_f32   = (float*)alloc((size_t)NT * DMODEL * 4);          // residual stream
    bf16*  x_bf    = (bf16*) alloc((size_t)NT * DMODEL * 2);
    bf16*  emb_bf  = (bf16*) alloc((size_t)VOCAB * DMODEL * 2);
    float* xp      = (float*)alloc((size_t)NT * 2 * DINNER * 4);      // W_in output
    bf16*  sx_bf   = (bf16*) alloc((size_t)NT * DINNER * 2);          // silu(x_ssm)
    float* z_f32   = (float*)alloc((size_t)NT * DINNER * 4);          // silu(z)
    float* sp      = (float*)alloc((size_t)NT * 1056 * 4);            // W_x output
    bf16*  dtp_bf  = (bf16*) alloc((size_t)NT * DINNER * 2);          // sp[:, :1024] bf16
    float* dtraw   = (float*)alloc((size_t)NT * DINNER * 4);          // -> dt in place
    bf16*  y_bf    = (bf16*) alloc((size_t)NT * DINNER * 2);          // scan output
    float* outb    = (float*)alloc((size_t)NT * DMODEL * 4);          // W_out output
    bf16*  wt      = (bf16*) alloc((size_t)DINNER * 1056 * 2);        // transposed weight

    const dim3 cvB(256);
    const int  cvG = 4096;       // grid-stride converters

    // Embed + bf16 copy of emb (emb is (V,D) row-major == N x K k-contig already)
    embed_kernel<<<dim3((NT * DMODEL + 255) / 256), cvB, 0, stream>>>(ids, emb, x_f32, x_bf);
    conv_strided_kernel<<<cvG, cvB, 0, stream>>>(emb, emb_bf, DMODEL, DMODEL,
                                                 (size_t)VOCAB * DMODEL);

    auto gemm = [&](const bf16* A, const bf16* Bt, float* C, int M, int N, int K) {
        wmma_gemm_bf16<<<dim3((N + 255) / 256, M / 32), dim3(128), 0, stream>>>(A, Bt, C, M, N, K);
    };

    for (int l = 0; l < NLAYER; ++l) {
        const float* W_in_l  = W_in  + (size_t)l * DMODEL * 2 * DINNER;
        const float* W_x_l   = W_x   + (size_t)l * DINNER * 1056;
        const float* W_dt_l  = W_dt  + (size_t)l * DINNER * DINNER;
        const float* b_dt_l  = b_dt  + (size_t)l * DINNER;
        const float* A_log_l = A_log + (size_t)l * DINNER * DSTATE;
        const float* D_par_l = D_par + (size_t)l * DINNER;
        const float* W_out_l = W_out + (size_t)l * DINNER * DMODEL;
        const float* ln_w_l  = ln_w  + (size_t)l * DMODEL;
        const float* ln_b_l  = ln_b  + (size_t)l * DMODEL;

        // xp = x @ W_in            (4096 x 512 x 2048)
        transpose_conv_kernel<<<cvG, cvB, 0, stream>>>(W_in_l, wt, DMODEL, 2 * DINNER);
        gemm(x_bf, wt, xp, NT, 2 * DINNER, DMODEL);

        split_silu_kernel<<<dim3((NT * DINNER + 255) / 256), cvB, 0, stream>>>(xp, sx_bf, z_f32);

        // sp = silu(x_ssm) @ W_x   (4096 x 1024 x 1056)
        transpose_conv_kernel<<<cvG, cvB, 0, stream>>>(W_x_l, wt, DINNER, 1056);
        gemm(sx_bf, wt, sp, NT, 1056, DINNER);

        // dtraw = dtp @ W_dt       (4096 x 1024 x 1024); dt = softplus(dtraw + b_dt)
        conv_strided_kernel<<<cvG, cvB, 0, stream>>>(sp, dtp_bf, 1056, DINNER,
                                                     (size_t)NT * DINNER);
        transpose_conv_kernel<<<cvG, cvB, 0, stream>>>(W_dt_l, wt, DINNER, DINNER);
        gemm(dtp_bf, wt, dtraw, NT, DINNER, DINNER);
        softplus_bias_kernel<<<dim3((NT * DINNER + 255) / 256), cvB, 0, stream>>>(dtraw, b_dt_l);

        // sequential selective scan -> y_bf
        scan_kernel<<<dim3(DINNER / 256, BATCH), dim3(256), 0, stream>>>(
            xp, sp, dtraw, z_f32, A_log_l, D_par_l, y_bf);

        // out = y @ W_out          (4096 x 1024 x 512)
        transpose_conv_kernel<<<cvG, cvB, 0, stream>>>(W_out_l, wt, DINNER, DMODEL);
        gemm(y_bf, wt, outb, NT, DMODEL, DINNER);

        // x = x + LN(out + x)
        layernorm_kernel<<<dim3(NT), dim3(256), 0, stream>>>(outb, x_f32, x_bf,
                                                             ln_w_l, ln_b_l, 1);
    }

    // final LN, then logits = x @ emb^T   (4096 x 512 x 32000)
    layernorm_kernel<<<dim3(NT), dim3(256), 0, stream>>>(nullptr, x_f32, x_bf,
                                                         fln_w, fln_b, 0);
    gemm(x_bf, emb_bf, (float*)d_out, NT, VOCAB, DMODEL);
}
